// VotePoseBackbone_10539849744921
// MI455X (gfx1250) — matrix operand order repaired and checked by reading, not compiled, not run on hardware
//
#include <hip/hip_runtime.h>
#include <stdint.h>

typedef _Float16 f16;
typedef __attribute__((ext_vector_type(8)))  _Float16 v8h;
typedef __attribute__((ext_vector_type(16))) _Float16 v16h;
typedef __attribute__((ext_vector_type(8)))  float    v8f;

// ---------------------------------------------------------------------------
// Utility: zero-fill (outputs accumulated with atomicMax must start at 0)
// ---------------------------------------------------------------------------
__global__ void zero_kernel(float* p, int n) {
    for (int t = blockIdx.x * blockDim.x + threadIdx.x; t < n;
         t += gridDim.x * blockDim.x)
        p[t] = 0.0f;
}

// ---------------------------------------------------------------------------
// Farthest point sampling. One block per batch. Points strided across threads,
// per-thread cached coords + running min-distance in registers.
// Matches jax scan: record `last` BEFORE update; argmax ties -> smallest index.
// ---------------------------------------------------------------------------
template <int PTS>
__global__ __launch_bounds__(1024) void fps_kernel(
    const float* __restrict__ xyz, int stride, int N, int npoint,
    int* __restrict__ outIdx) {
    const int b = blockIdx.x, tid = threadIdx.x, T = blockDim.x;
    const float* base = xyz + (size_t)b * N * stride;

    float px[PTS], py[PTS], pz[PTS], dist[PTS];
#pragma unroll
    for (int i = 0; i < PTS; ++i) {
        int p = tid + i * T;
        if (p < N) {
            px[i] = base[(size_t)p * stride + 0];
            py[i] = base[(size_t)p * stride + 1];
            pz[i] = base[(size_t)p * stride + 2];
            dist[i] = 1e10f;
        } else {
            px[i] = py[i] = pz[i] = 0.0f;
            dist[i] = -1e30f;  // never wins argmax
        }
    }

    __shared__ float cpt[3];
    __shared__ float redV[32];
    __shared__ int   redI[32];
    __shared__ int   lastSh;
    const int lane = tid & 31, wave = tid >> 5, nw = T >> 5;

    int last = 0;
    for (int it = 0; it < npoint; ++it) {
        if (tid == 0) outIdx[b * npoint + it] = last;
        if ((last % T) == tid) {
            int i = last / T;
            cpt[0] = px[i]; cpt[1] = py[i]; cpt[2] = pz[i];
        }
        __syncthreads();
        float cx = cpt[0], cy = cpt[1], cz = cpt[2];

        float bv = -1e30f; int bi = 0x7fffffff;
#pragma unroll
        for (int i = 0; i < PTS; ++i) {
            int p = tid + i * T;
            float dx = px[i] - cx, dy = py[i] - cy, dz = pz[i] - cz;
            float d = dx * dx + dy * dy + dz * dz;
            dist[i] = fminf(dist[i], d);
            if (dist[i] > bv || (dist[i] == bv && p < bi)) { bv = dist[i]; bi = p; }
        }
        // wave32 reduction
        for (int off = 16; off > 0; off >>= 1) {
            float ov = __shfl_down(bv, off, 32);
            int   oi = __shfl_down(bi, off, 32);
            if (ov > bv || (ov == bv && oi < bi)) { bv = ov; bi = oi; }
        }
        if (lane == 0) { redV[wave] = bv; redI[wave] = bi; }
        __syncthreads();
        if (wave == 0) {
            float v2 = (lane < nw) ? redV[lane] : -1e30f;
            int   i2 = (lane < nw) ? redI[lane] : 0x7fffffff;
            for (int off = 16; off > 0; off >>= 1) {
                float ov = __shfl_down(v2, off, 32);
                int   oi = __shfl_down(i2, off, 32);
                if (ov > v2 || (ov == v2 && oi < i2)) { v2 = ov; i2 = oi; }
            }
            if (lane == 0) lastSh = i2;
        }
        __syncthreads();
        last = lastSh;
    }
}

// ---------------------------------------------------------------------------
// Gather xyz coordinates at sampled indices -> [totalCenters, 3]
// ---------------------------------------------------------------------------
__global__ void gather_xyz_kernel(const float* __restrict__ xyz, int stride,
                                  int N, const int* __restrict__ idx,
                                  int npoint, float* __restrict__ out,
                                  int total) {
    int t = blockIdx.x * blockDim.x + threadIdx.x;
    if (t >= total) return;
    int b = t / npoint;
    int p = idx[t];
    const float* s = xyz + ((size_t)b * N + p) * stride;
    out[t * 3 + 0] = s[0];
    out[t * 3 + 1] = s[1];
    out[t * 3 + 2] = s[2];
}

// ---------------------------------------------------------------------------
// Ball query: one wave32 per center; ballot-compacted first-nsample in-radius
// indices (ascending index order == jax top_k(-cand) smallest indices).
// Pad with first found; if none found, pad with N-1 (matches clip semantics).
// ---------------------------------------------------------------------------
__global__ __launch_bounds__(256) void ball_query_kernel(
    const float* __restrict__ xyz, int stride, int N,
    const float* __restrict__ newXyz, int npoint, int nsample, float r2,
    int* __restrict__ gIdx, int totalCenters) {
    const int wave = threadIdx.x >> 5, lane = threadIdx.x & 31;
    const int center = blockIdx.x * (blockDim.x >> 5) + wave;
    if (center >= totalCenters) return;
    const int b = center / npoint;
    const float cx = newXyz[center * 3 + 0];
    const float cy = newXyz[center * 3 + 1];
    const float cz = newXyz[center * 3 + 2];
    const float* base = xyz + (size_t)b * N * stride;
    int* out = gIdx + (size_t)center * nsample;

    int count = 0, firstIdx = -1;
    for (int p0 = 0; p0 < N && count < nsample; p0 += 32) {
        int p = p0 + lane;
        // prefetch next chunk of coordinates (global_prefetch_b8)
        if (p0 + 32 < N)
            __builtin_prefetch(base + (size_t)(p + 32) * stride, 0, 1);
        bool pred = false;
        if (p < N) {
            float dx = base[(size_t)p * stride + 0] - cx;
            float dy = base[(size_t)p * stride + 1] - cy;
            float dz = base[(size_t)p * stride + 2] - cz;
            pred = (dx * dx + dy * dy + dz * dz) <= r2;
        }
        uint32_t mask = (uint32_t)__ballot(pred);
        if (firstIdx < 0 && mask) firstIdx = p0 + __ffs((int)mask) - 1;
        int pos = __popc(mask & ((lane == 0) ? 0u : ((1u << lane) - 1u)));
        if (pred && (count + pos) < nsample) out[count + pos] = p;
        count += __popc(mask);
    }
    if (count > nsample) count = nsample;
    int fill = (firstIdx < 0) ? (N - 1) : firstIdx;
    for (int k = count + lane; k < nsample; k += 32) out[k] = fill;
}

// ---------------------------------------------------------------------------
// WMMA MLP layer over a 64-row tile held in LDS (f16, row stride = Kp).
//   Y = relu( X @ fold(W)^T + fold(b) )   with BN folded: s = gamma/sqrt(1+eps)
// Weights staged into LDS in 32-K chunks and stored PRE-SWIZZLED into the
// wave32 B-fragment layout: element (k, n) -> Wlds[((k>>4)*Nout + n)*16 + (k&15)]
// so each lane's B operand is one contiguous 32-byte run (2x ds_load_b128).
// A operand is two aligned 16-byte runs per lane (2x ds_load_b128).
// The wave index is scalarized with readfirstlane so tile dispatch is pure
// SALU control flow: EXEC stays all-ones at every v_wmma (ISA 7.12 requirement)
// with no saveexec predication around the matrix ops.
// Output modes: Ylds (f16 LDS), gmax (global int-as-float atomic maxpool),
// gout (global fp32 rows).
// Fragment layouts use k = 16*(lane/16) + s in both A and B (wave32, 16-bit A 16x32, B 32x16).
// ---------------------------------------------------------------------------
__device__ inline void mlp_layer64(
    const f16* __restrict__ X, int Kp, int Cin,
    const float* __restrict__ W, const float* __restrict__ bias,
    const float* __restrict__ gamma, const float* __restrict__ beta, int Nout,
    f16* Wlds, float* biasLds, f16* Ylds, float* gmax, float* gout,
    int rowBase, int outStride) {
    const int tid = threadIdx.x;
    const int lane = tid & 31;
    // wave-uniform scalar: lets the compiler use SALU branches for tile bounds
    const int wave = __builtin_amdgcn_readfirstlane(tid >> 5);
    const int numWaves = (int)(blockDim.x >> 5);
    const int numNt = Nout >> 4;
    const int numT = 4 * numNt;  // 64 rows -> 4 M tiles
    const float rsq = rsqrtf(1.0f + 1e-5f);

    for (int n = tid; n < Nout; n += blockDim.x)
        biasLds[n] = bias[n] * (gamma[n] * rsq) + beta[n];

    v8f acc[4] = {};
    const int hiHalf = lane >> 4;
    const int numKt = Kp >> 5;

    for (int kt = 0; kt < numKt; ++kt) {
        __syncthreads();  // previous chunk consumed / bias visible
        for (int idx = tid; idx < 32 * Nout; idx += blockDim.x) {
            int kloc = idx / Nout, n = idx - kloc * Nout;
            int kg = (kt << 5) + kloc;
            float wv = (kg < Cin) ? W[(size_t)n * Cin + kg] * (gamma[n] * rsq)
                                  : 0.0f;
            // swizzled B-fragment store
            Wlds[(((kloc >> 4) * Nout + n) << 4) + (kloc & 15)] = (f16)wv;
        }
        __syncthreads();
#pragma unroll
        for (int i = 0; i < 4; ++i) {
            int t = wave + i * numWaves;
            if (t >= numT) break;  // scalar condition -> s_cbranch, EXEC intact
            int mt = t / numNt;
            int mBase = mt << 4, nBase = (t - mt * numNt) << 4;
            // A fragment: rows of X, two contiguous 8-half chunks per lane
            const f16* Xrow = X + (size_t)(mBase + (lane & 15)) * Kp + (kt << 5);
            v8h alo = *(const v8h*)(Xrow + (hiHalf << 4));
            v8h ahi = *(const v8h*)(Xrow + (hiHalf << 4) + 8);
            v16h a = __builtin_shufflevector(alo, ahi, 0, 1, 2, 3, 4, 5, 6, 7,
                                             8, 9, 10, 11, 12, 13, 14, 15);
            // B fragment: one contiguous 32-byte run per lane
            const f16* Wp =
                Wlds + (((hiHalf * Nout) + nBase + (lane & 15)) << 4);
            v16h bf = *(const v16h*)Wp;
            acc[i] = __builtin_amdgcn_wmma_f32_16x16x32_f16(
                false, a, false, bf, (short)0, acc[i], false, false);
        }
    }
    __syncthreads();
    // epilogue: bias + relu + dispatch
#pragma unroll
    for (int i = 0; i < 4; ++i) {
        int t = wave + i * numWaves;
        if (t >= numT) break;
        int mt = t / numNt;
        int mBase = mt << 4, nBase = (t - mt * numNt) << 4;
        int n = nBase + (lane & 15);
        float bb = biasLds[n];
        float colmax = 0.0f;
#pragma unroll
        for (int r = 0; r < 8; ++r) {
            int m = mBase + r + (hiHalf << 3);
            float val = fmaxf(acc[i][r] + bb, 0.0f);
            if (Ylds) Ylds[(size_t)m * Nout + n] = (f16)val;
            if (gout) gout[(size_t)(rowBase + m) * outStride + n] = val;
            colmax = fmaxf(colmax, val);
        }
        if (gmax) {
            float o = __shfl_xor(colmax, 16, 32);
            colmax = fmaxf(colmax, o);
            if (lane < 16)
                atomicMax((int*)&gmax[n], __float_as_int(colmax));  // vals >= 0
        }
    }
    __syncthreads();
}

// ---------------------------------------------------------------------------
// SA module grouped MLP + maxpool. One workgroup per (center, 64-row chunk).
// Row channels: [ (xyz - center)/radius (3) | features (F) | zero pad -> K0P ]
// ---------------------------------------------------------------------------
template <int K0P>
__global__ __launch_bounds__(256) void sa_mlp_kernel(
    const float* __restrict__ xyz, int xyzStride, int Nsrc,
    const float* __restrict__ feat, int featStride, int F,
    const float* __restrict__ newXyz, const int* __restrict__ gIdx,
    float radius, int npoint, int nsample,
    const float* W1, const float* b1, const float* g1, const float* be1,
    const float* W2, const float* b2, const float* g2, const float* be2,
    const float* W3, const float* b3, const float* g3, const float* be3,
    float* __restrict__ outFeat /* [totalCenters,128] pre-zeroed */) {
    __shared__ f16 X0[64 * K0P];
    __shared__ f16 X1[64 * 64];
    __shared__ f16 Wlds[32 * 128];
    __shared__ float biasLds[128];

    const int chunks = nsample >> 6;
    const int chunk = blockIdx.x % chunks;
    const int center = blockIdx.x / chunks;
    const int b = center / npoint;
    const int tid = threadIdx.x;
    const int Cin = 3 + F;

    const float c0 = newXyz[center * 3 + 0];
    const float c1 = newXyz[center * 3 + 1];
    const float c2 = newXyz[center * 3 + 2];
    const float invR = 1.0f / radius;
    const int* gi = gIdx + (size_t)center * nsample + (size_t)chunk * 64;

    for (int idx = tid; idx < 64 * K0P; idx += blockDim.x) {
        int s = idx / K0P, k = idx - s * K0P;
        int pi = gi[s];
        float v = 0.0f;
        if (k < 3) {
            float c = (k == 0) ? c0 : (k == 1) ? c1 : c2;
            v = (xyz[((size_t)b * Nsrc + pi) * xyzStride + k] - c) * invR;
        } else if (k < Cin) {
            v = feat[((size_t)b * Nsrc + pi) * featStride + (k - 3)];
        }
        X0[idx] = (f16)v;
    }
    __syncthreads();

    // layer 1: (64 x K0P) -> (64 x 64)
    mlp_layer64(X0, K0P, Cin, W1, b1, g1, be1, 64, Wlds, biasLds,
                X1, nullptr, nullptr, 0, 0);
    // layer 2: (64 x 64) -> (64 x 64), reuse X0 buffer
    mlp_layer64(X1, 64, 64, W2, b2, g2, be2, 64, Wlds, biasLds,
                X0, nullptr, nullptr, 0, 0);
    // layer 3: (64 x 64) -> (64 x 128), fused maxpool into global atomics
    mlp_layer64(X0, 64, 64, W3, b3, g3, be3, 128, Wlds, biasLds,
                nullptr, outFeat + (size_t)center * 128, nullptr, 0, 0);
}

// ---------------------------------------------------------------------------
// FP: 3-NN inverse-distance interpolation of sa2 features onto sa1 points.
// ---------------------------------------------------------------------------
__global__ __launch_bounds__(256) void fp_interp_kernel(
    const float* __restrict__ unknown /*[B,nUnk,3]*/,
    const float* __restrict__ known /*[B,nKnown,3]*/,
    const float* __restrict__ feat2 /*[B,nKnown,C]*/,
    float* __restrict__ interp /*[B,nUnk,C]*/, int nUnk, int nKnown, int C,
    int total) {
    int t = blockIdx.x * blockDim.x + threadIdx.x;
    if (t >= total) return;
    int b = t / nUnk;
    float ux = unknown[t * 3 + 0], uy = unknown[t * 3 + 1], uz = unknown[t * 3 + 2];
    float bd0 = 1e30f, bd1 = 1e30f, bd2 = 1e30f;
    int bi0 = 0, bi1 = 0, bi2 = 0;
    const float* kb = known + (size_t)b * nKnown * 3;
    for (int k = 0; k < nKnown; ++k) {
        float dx = ux - kb[k * 3 + 0];
        float dy = uy - kb[k * 3 + 1];
        float dz = uz - kb[k * 3 + 2];
        float d = dx * dx + dy * dy + dz * dz;
        if (d < bd0) {
            bd2 = bd1; bi2 = bi1; bd1 = bd0; bi1 = bi0; bd0 = d; bi0 = k;
        } else if (d < bd1) {
            bd2 = bd1; bi2 = bi1; bd1 = d; bi1 = k;
        } else if (d < bd2) {
            bd2 = d; bi2 = k;
        }
    }
    float w0 = 1.0f / (bd0 + 1e-8f);
    float w1 = 1.0f / (bd1 + 1e-8f);
    float w2 = 1.0f / (bd2 + 1e-8f);
    float ws = w0 + w1 + w2;
    w0 /= ws; w1 /= ws; w2 /= ws;
    const float* f2 = feat2 + (size_t)b * nKnown * C;
    float* o = interp + (size_t)t * C;
    for (int c = 0; c < C; ++c)
        o[c] = w0 * f2[(size_t)bi0 * C + c] + w1 * f2[(size_t)bi1 * C + c] +
               w2 * f2[(size_t)bi2 * C + c];
}

// ---------------------------------------------------------------------------
// FP MLP: rows = sa1 points (256/batch, 64-row chunks), Cin = 256
// (interp 128 | sa1_feat 128), layers 256->128->128, fp32 out rows.
// ---------------------------------------------------------------------------
__global__ __launch_bounds__(256) void fp_mlp_kernel(
    const float* __restrict__ interp /*[B,256,128]*/,
    const float* __restrict__ feat1 /*[B,256,128]*/,
    const float* W1, const float* b1, const float* g1, const float* be1,
    const float* W2, const float* b2, const float* g2, const float* be2,
    float* __restrict__ out /*[B,256,128]*/) {
    __shared__ f16 X0[64 * 256];
    __shared__ f16 X1[64 * 128];
    __shared__ f16 Wlds[32 * 128];
    __shared__ float biasLds[128];

    const int chunk = blockIdx.x & 3;
    const int b = blockIdx.x >> 2;
    const int rowBase = chunk * 64;
    const int tid = threadIdx.x;

    for (int idx = tid; idx < 64 * 256; idx += blockDim.x) {
        int s = idx >> 8, k = idx & 255;
        int i = rowBase + s;
        float v = (k < 128) ? interp[((size_t)b * 256 + i) * 128 + k]
                            : feat1[((size_t)b * 256 + i) * 128 + (k - 128)];
        X0[idx] = (f16)v;
    }
    __syncthreads();

    mlp_layer64(X0, 256, 256, W1, b1, g1, be1, 128, Wlds, biasLds,
                X1, nullptr, nullptr, 0, 0);
    mlp_layer64(X1, 128, 128, W2, b2, g2, be2, 128, Wlds, biasLds,
                nullptr, nullptr, out + (size_t)b * 256 * 128, rowBase, 128);
}

// ---------------------------------------------------------------------------
// Host-side launch
// ---------------------------------------------------------------------------
extern "C" void kernel_launch(void* const* d_in, const int* in_sizes, int n_in,
                              void* d_out, int out_size, void* d_ws,
                              size_t ws_size, hipStream_t stream) {
    (void)in_sizes; (void)n_in; (void)out_size; (void)ws_size;
    constexpr int B = 8, N = 16384, F = 32;
    constexpr int NP1 = 256, NS1 = 256;
    constexpr int NP2 = 128, NS2 = 64;

    const float* pc = (const float*)d_in[0];
    // per-layer dict insertion order: W, b, gamma, beta
    auto P = [&](int i) { return (const float*)d_in[i]; };
    // sa1: d_in[1..12], sa2: d_in[13..24], fp1: d_in[25..32]
    const int S1 = 1, S2 = 13, FP = 25;

    // workspace carve-up
    auto up = [](size_t x) { return (x + 255) & ~(size_t)255; };
    char* w = (char*)d_ws;
    size_t off = 0;
    int*   fps1 = (int*)(w + off);   off = up(off + (size_t)B * NP1 * 4);
    float* nx1  = (float*)(w + off); off = up(off + (size_t)B * NP1 * 3 * 4);
    int*   gi1  = (int*)(w + off);   off = up(off + (size_t)B * NP1 * NS1 * 4);
    float* sf1  = (float*)(w + off); off = up(off + (size_t)B * NP1 * 128 * 4);
    int*   fps2 = (int*)(w + off);   off = up(off + (size_t)B * NP2 * 4);
    float* nx2  = (float*)(w + off); off = up(off + (size_t)B * NP2 * 3 * 4);
    int*   gi2  = (int*)(w + off);   off = up(off + (size_t)B * NP2 * NS2 * 4);
    float* sf2  = (float*)(w + off); off = up(off + (size_t)B * NP2 * 128 * 4);
    float* itp  = (float*)(w + off); off = up(off + (size_t)B * NP1 * 128 * 4);
    float* outF = (float*)d_out;

    // zero the maxpool accumulators (atomicMax targets)
    zero_kernel<<<256, 256, 0, stream>>>(sf1, B * NP1 * 128);
    zero_kernel<<<256, 256, 0, stream>>>(sf2, B * NP2 * 128);

    // ---- SA1 ----
    fps_kernel<16><<<B, 1024, 0, stream>>>(pc, 3 + F, N, NP1, fps1);
    gather_xyz_kernel<<<(B * NP1 + 255) / 256, 256, 0, stream>>>(
        pc, 3 + F, N, fps1, NP1, nx1, B * NP1);
    ball_query_kernel<<<(B * NP1) / 8, 256, 0, stream>>>(
        pc, 3 + F, N, nx1, NP1, NS1, 0.2f * 0.2f, gi1, B * NP1);
    sa_mlp_kernel<64><<<B * NP1 * (NS1 / 64), 256, 0, stream>>>(
        pc, 3 + F, N, pc + 3, 3 + F, F, nx1, gi1, 0.2f, NP1, NS1,
        P(S1 + 0), P(S1 + 1), P(S1 + 2), P(S1 + 3),
        P(S1 + 4), P(S1 + 5), P(S1 + 6), P(S1 + 7),
        P(S1 + 8), P(S1 + 9), P(S1 + 10), P(S1 + 11), sf1);

    // ---- SA2 ----
    fps_kernel<16><<<B, 1024, 0, stream>>>(nx1, 3, NP1, NP2, fps2);
    gather_xyz_kernel<<<(B * NP2 + 255) / 256, 256, 0, stream>>>(
        nx1, 3, NP1, fps2, NP2, nx2, B * NP2);
    ball_query_kernel<<<(B * NP2) / 8, 256, 0, stream>>>(
        nx1, 3, NP1, nx2, NP2, NS2, 0.4f * 0.4f, gi2, B * NP2);
    sa_mlp_kernel<160><<<B * NP2 * (NS2 / 64), 256, 0, stream>>>(
        nx1, 3, NP1, sf1, 128, 128, nx2, gi2, 0.4f, NP2, NS2,
        P(S2 + 0), P(S2 + 1), P(S2 + 2), P(S2 + 3),
        P(S2 + 4), P(S2 + 5), P(S2 + 6), P(S2 + 7),
        P(S2 + 8), P(S2 + 9), P(S2 + 10), P(S2 + 11), sf2);

    // ---- FP1 ----
    fp_interp_kernel<<<(B * NP1 + 255) / 256, 256, 0, stream>>>(
        nx1, nx2, sf2, itp, NP1, NP2, 128, B * NP1);
    fp_mlp_kernel<<<B * (NP1 / 64), 256, 0, stream>>>(
        itp, sf1,
        P(FP + 0), P(FP + 1), P(FP + 2), P(FP + 3),
        P(FP + 4), P(FP + 5), P(FP + 6), P(FP + 7), outF);
}
